// Refinement_57200374448506
// MI455X (gfx1250) — matrix operand order repaired and checked by reading, not compiled, not run on hardware
//
#include <hip/hip_runtime.h>
#include <hip/hip_bf16.h>

// ---------------------------------------------------------------------------
// CDNA5 (gfx1250) sparse-conv refinement network.
// All GEMM work runs through v_wmma_f32_16x16x32_f16 (wave32 tiles of 16x32).
// Features are kept in fp16 (padded to 32/96 channels, +1 zero sentinel row),
// weights are repacked once per call into the per-lane B-fragment layout and
// staged in LDS per workgroup (ds_load feeds the WMMA B operand).
// ---------------------------------------------------------------------------

typedef __attribute__((ext_vector_type(16))) _Float16 v16h;
typedef __attribute__((ext_vector_type(8)))  _Float16 v8h;
typedef __attribute__((ext_vector_type(8)))  float    v8f;

#define TPB 256
#define BN_EPS 1e-4f

// ---------------------------------------------------------------------------
// Weight repack: W [K, Cin, 32] f32  ->  Bpack halves laid out as
// [k][chunk][ntile][lane 0..31][idx 0..15]
// B-fragment order chosen here (16-bit B, 32x16; contiguous k, A uses the same): lanes 0-15 hold K=0..15, lanes 16-31
// hold K=16..31; within a lane VGPR r holds K pair (2r,2r+1) -> K = half*16+idx.
// Zero-pads Cin up to chunks*32 (handles Cin=16 -> 32).
// ---------------------------------------------------------------------------
__global__ void pack_w_kernel(const float* __restrict__ W, _Float16* __restrict__ dst,
                              int K, int Cin, int chunks) {
    int total = K * chunks * 1024;            // 2 ntiles * 32 lanes * 16 idx
    int gid = blockIdx.x * blockDim.x + threadIdx.x;
    if (gid >= total) return;
    int idx   = gid & 15;
    int lane  = (gid >> 4) & 31;
    int nt    = (gid >> 9) & 1;
    int chunk = (gid >> 10) % chunks;
    int k     = gid / (chunks << 10);
    int n     = lane & 15;
    int halfL = lane >> 4;
    int kin   = chunk * 32 + halfL * 16 + idx;
    float v   = (kin < Cin) ? W[((size_t)k * Cin + kin) * 32 + nt * 16 + n] : 0.f;
    dst[gid] = (_Float16)v;
}

// f32 [M,Cin] -> f16 [(M+1), Cpad], zero-padded columns, zero sentinel row M.
__global__ void feats_to_h_kernel(const float* __restrict__ X, _Float16* __restrict__ dst,
                                  int M, int Cin, int Cpad) {
    long long total = (long long)(M + 1) * Cpad;
    long long gid = (long long)blockIdx.x * blockDim.x + threadIdx.x;
    if (gid >= total) return;
    int row = (int)(gid / Cpad), c = (int)(gid % Cpad);
    float v = (row < M && c < Cin) ? X[(size_t)row * Cin + c] : 0.f;
    dst[gid] = (_Float16)v;
}

// ---------------------------------------------------------------------------
// BatchNorm statistics: stats[0:C] = sum, stats[C:2C] = sum of squares.
// stats must be zeroed before launch (hipMemsetAsync in kernel_launch).
// Lane c maps to a distinct LDS bank -> shared atomics are conflict-free.
// ---------------------------------------------------------------------------
__global__ void bn_stats_kernel(const _Float16* __restrict__ X, float* __restrict__ stats,
                                int M, int C) {
    __shared__ float ssum[96];
    __shared__ float ssq[96];
    for (int c = threadIdx.x; c < C; c += blockDim.x) { ssum[c] = 0.f; ssq[c] = 0.f; }
    __syncthreads();
    long long total = (long long)M * C;
    for (long long gid = (long long)blockIdx.x * blockDim.x + threadIdx.x; gid < total;
         gid += (long long)gridDim.x * blockDim.x) {
        float v = (float)X[gid];
        int c = (int)(gid % C);
        atomicAdd(&ssum[c], v);
        atomicAdd(&ssq[c], v * v);
    }
    __syncthreads();
    for (int c = threadIdx.x; c < C; c += blockDim.x) {
        atomicAdd(&stats[c],     ssum[c]);
        atomicAdd(&stats[C + c], ssq[c]);
    }
}

// y = relu((x-mean)*rsqrt(var+eps)*g + b), writes (M+1) rows (sentinel row = 0).
__global__ void bn_apply_kernel(const _Float16* __restrict__ X, _Float16* __restrict__ Y,
                                const float* __restrict__ stats,
                                const float* __restrict__ g, const float* __restrict__ b,
                                int M, int C) {
    long long total = (long long)(M + 1) * C;
    long long gid = (long long)blockIdx.x * blockDim.x + threadIdx.x;
    if (gid >= total) return;
    int row = (int)(gid / C), c = (int)(gid % C);
    if (row >= M) { Y[gid] = (_Float16)0.f; return; }
    float invM  = 1.f / (float)M;
    float mean  = stats[c] * invM;
    float var   = stats[C + c] * invM - mean * mean;
    float v = ((float)X[gid] - mean) * rsqrtf(var + BN_EPS) * g[c] + b[c];
    Y[gid] = (_Float16)fmaxf(v, 0.f);
}

__global__ void add_kernel(_Float16* __restrict__ Y, const _Float16* __restrict__ X,
                           long long n) {
    long long i = (long long)blockIdx.x * blockDim.x + threadIdx.x;
    if (i < n) Y[i] = (_Float16)((float)Y[i] + (float)X[i]);
}

// H[row, 0:32]=x0[row]; [32:64]=y0[parent0[row]]; [64:96]=z0[parent1[parent0[row]]]
__global__ void concat3_kernel(const _Float16* __restrict__ X0, const _Float16* __restrict__ Y0,
                               const _Float16* __restrict__ Z0,
                               const int* __restrict__ parent0, const int* __restrict__ parent1,
                               _Float16* __restrict__ H, int M0) {
    long long total = (long long)M0 * 96;
    long long gid = (long long)blockIdx.x * blockDim.x + threadIdx.x;
    if (gid >= total) return;
    int row = (int)(gid / 96), c = (int)(gid % 96);
    _Float16 v;
    if (c < 32) v = X0[(size_t)row * 32 + c];
    else {
        int p = parent0[row];
        if (c < 64) v = Y0[(size_t)p * 32 + (c - 32)];
        else        v = Z0[(size_t)parent1[p] * 32 + (c - 64)];
    }
    H[gid] = v;
}

// ---------------------------------------------------------------------------
// Gather-GEMM-scatter conv via WMMA, statically specialized on (CHUNKS, K).
//   Xh  : [(rows+1), CHUNKS*32] fp16, zero row at index (sentinel >> shift)
//   Bp  : packed weights [K][CHUNKS][2][32][16] halves, staged into LDS
//   nbr : [K, Mout] int32 rulebook (sentinel value == `sentinel`)
// Each wave computes a 16-row x 32-col tile; rulebook indices are preloaded
// into registers, gather rows are prefetched one offset ahead, and the fully
// unrolled K x CHUNKS loop issues 2 v_wmma_f32_16x16x32_f16 per step with the
// B operand fed from LDS (ds_load_b128).
// ---------------------------------------------------------------------------
template <int CHUNKS, int K>
__global__ void __launch_bounds__(256)
conv_wmma_kernel(const _Float16* __restrict__ Xh, const _Float16* __restrict__ Bp,
                 const int* __restrict__ nbr, _Float16* __restrict__ Out,
                 int Mout, int sentinel, int shift) {
    constexpr int BHALVES = K * CHUNKS * 1024;
    extern __shared__ _Float16 sB[];

    // cooperative staging of packed weights into LDS (16B vectors)
    for (int i = threadIdx.x; i < BHALVES / 8; i += blockDim.x)
        ((v8h*)sB)[i] = ((const v8h*)Bp)[i];
    __syncthreads();

    int wave = blockIdx.x * (blockDim.x >> 5) + (threadIdx.x >> 5);   // wave-uniform
    int numTiles = (Mout + 15) >> 4;
    if (wave >= numTiles) return;                                     // EXEC stays all-1s
    int lane  = threadIdx.x & 31;
    int m     = lane & 15;
    int halfL = lane >> 4;
    int row   = wave * 16 + m;
    constexpr int Cpad = CHUNKS * 32;

    // preload the whole rulebook column for this output row
    int nbrs[K];
#pragma unroll
    for (int k = 0; k < K; ++k)
        nbrs[k] = (row < Mout) ? nbr[(size_t)k * Mout + row] : sentinel;

    v8f acc0 = {};
    v8f acc1 = {};

#pragma unroll
    for (int k = 0; k < K; ++k) {
        const _Float16* src = Xh + (size_t)(nbrs[k] >> shift) * Cpad;
        if (k + 1 < K)   // prefetch next offset's gather row (global_prefetch_b8)
            __builtin_prefetch(Xh + (size_t)(nbrs[k + 1] >> shift) * Cpad, 0, 0);
        const _Float16* bk = sB + (size_t)k * CHUNKS * 1024 + (size_t)lane * 16;
#pragma unroll
        for (int c = 0; c < CHUNKS; ++c) {
            // A fragment (16-bit A 16x32): lane half selects K = h*16 .. h*16+15 (matches B pack)
            v8h lo = *(const v8h*)(src + c * 32 + halfL * 16);
            v8h hi = *(const v8h*)(src + c * 32 + halfL * 16 + 8);
            v16h a;
#pragma unroll
            for (int i = 0; i < 8; ++i) { a[i] = lo[i]; a[8 + i] = hi[i]; }
            v16h b0 = *(const v16h*)(bk + c * 1024);         // ntile 0
            v16h b1 = *(const v16h*)(bk + c * 1024 + 512);   // ntile 1
            acc0 = __builtin_amdgcn_wmma_f32_16x16x32_f16(false, a, false, b0,
                                                          (short)0, acc0, false, false);
            acc1 = __builtin_amdgcn_wmma_f32_16x16x32_f16(false, a, false, b1,
                                                          (short)0, acc1, false, false);
        }
    }
    // D layout: VGPR r -> M = r + 8*halfL, N = lane%16
#pragma unroll
    for (int r = 0; r < 8; ++r) {
        int ro = wave * 16 + r + halfL * 8;
        if (ro < Mout) {
            Out[(size_t)ro * 32 + m]      = (_Float16)acc0[r];
            Out[(size_t)ro * 32 + 16 + m] = (_Float16)acc1[r];
        }
    }
}

// ---------------------------------------------------------------------------
// Final head: bn-relu(xn) -> occ/sdf linears, sigmoid mask, octree locs_next,
// writes the 4 flattened outputs into d_out (locs | feats_out | out | mask).
// ---------------------------------------------------------------------------
__global__ void head_kernel(const _Float16* __restrict__ XN, const float* __restrict__ stats,
                            const float* __restrict__ g, const float* __restrict__ b,
                            const float* __restrict__ Wlin, const float* __restrict__ blin,
                            const float* __restrict__ Wsdf, const float* __restrict__ bsdf,
                            const int* __restrict__ locs, float* __restrict__ out, int Mn) {
    int i = blockIdx.x * blockDim.x + threadIdx.x;
    if (i >= Mn) return;
    float invM = 1.f / (float)Mn;
    float xn[32];
    float occ = blin[0], sdf = bsdf[0];
#pragma unroll
    for (int c = 0; c < 32; ++c) {
        float mean = stats[c] * invM;
        float var  = stats[32 + c] * invM - mean * mean;
        float v = ((float)XN[(size_t)i * 32 + c] - mean) * rsqrtf(var + BN_EPS) * g[c] + b[c];
        v = fmaxf(v, 0.f);
        xn[c] = v;
        occ += v * Wlin[c];
        sdf += v * Wsdf[c];
    }
    float sg = 1.f / (1.f + expf(-occ));
    float maskf = (sg > 0.5f) ? 1.f : 0.f;

    float* locsOut  = out;
    float* featsOut = out + (size_t)Mn * 4;
    float* outOut   = out + (size_t)Mn * 38;
    float* maskOut  = out + (size_t)Mn * 40;

    int parent = i >> 3, cidx = i & 7;
    int ox = (cidx >> 2) & 1, oy = (cidx >> 1) & 1, oz = cidx & 1;
    locsOut[(size_t)i * 4 + 0] = (float)(locs[parent * 4 + 0] * 2 + ox);
    locsOut[(size_t)i * 4 + 1] = (float)(locs[parent * 4 + 1] * 2 + oy);
    locsOut[(size_t)i * 4 + 2] = (float)(locs[parent * 4 + 2] * 2 + oz);
    locsOut[(size_t)i * 4 + 3] = (float)(locs[parent * 4 + 3]);
#pragma unroll
    for (int c = 0; c < 32; ++c) featsOut[(size_t)i * 34 + c] = xn[c] * maskf;
    featsOut[(size_t)i * 34 + 32] = occ * maskf;
    featsOut[(size_t)i * 34 + 33] = sdf * maskf;
    outOut[(size_t)i * 2 + 0] = occ;
    outOut[(size_t)i * 2 + 1] = sdf;
    maskOut[i] = maskf;
}

// ---------------------------------------------------------------------------

extern "C" void kernel_launch(void* const* d_in, const int* in_sizes, int n_in,
                              void* d_out, int out_size, void* d_ws, size_t ws_size,
                              hipStream_t stream) {
    const int*   locs  = (const int*)d_in[0];
    const float* feats = (const float*)d_in[1];
    const float* W1    = (const float*)d_in[2];
    const float* Wres  = (const float*)d_in[3];
    const float* Wdown = (const float*)d_in[4];
    const float* Wn1   = (const float*)d_in[5];
    const float* bn_g  = (const float*)d_in[6];
    const float* bn_b  = (const float*)d_in[7];
    const float* g3    = (const float*)d_in[8];
    const float* b3    = (const float*)d_in[9];
    const float* Wlin  = (const float*)d_in[10];
    const float* blin  = (const float*)d_in[11];
    const float* Wsdf  = (const float*)d_in[12];
    const float* bsdf  = (const float*)d_in[13];
    const int* rb0     = (const int*)d_in[14];
    const int* rb1     = (const int*)d_in[15];
    const int* rb2     = (const int*)d_in[16];
    const int* rbn     = (const int*)d_in[17];
    const int* child0  = (const int*)d_in[18];
    const int* child1  = (const int*)d_in[19];
    const int* parent0 = (const int*)d_in[20];
    const int* parent1 = (const int*)d_in[21];

    const int M0 = in_sizes[14] / 27;
    const int M1 = in_sizes[15] / 27;
    const int M2 = in_sizes[16] / 27;
    const int Mn = in_sizes[17] / 27;

    // allow >64KB dynamic LDS for the 3-chunk instantiation (162 KB; WGP has 320 KB)
    hipFuncSetAttribute((const void*)conv_wmma_kernel<3, 27>,
                        hipFuncAttributeMaxDynamicSharedMemorySize, 3 * 27 * 1024 * 2);

    // --- workspace arena ---
    char* ws = (char*)d_ws;
    size_t off = 0;
    auto alloc = [&](size_t bytes) -> void* {
        void* p = ws + off;
        off = (off + bytes + 255) & ~(size_t)255;
        return p;
    };

    float* stats = (float*)alloc(10 * 192 * sizeof(float));   // 10 slots x (2*96)
    auto st = [&](int i) { return stats + (size_t)i * 192; };

    _Float16* W1p = (_Float16*)alloc((size_t)27 * 1024 * 2);
    _Float16* Wresp[6];
    for (int i = 0; i < 6; ++i) Wresp[i] = (_Float16*)alloc((size_t)27 * 1024 * 2);
    _Float16* Wdownp[2];
    for (int i = 0; i < 2; ++i) Wdownp[i] = (_Float16*)alloc((size_t)8 * 1024 * 2);
    _Float16* Wn1p = (_Float16*)alloc((size_t)27 * 3 * 1024 * 2);

    auto fbuf = [&](int rows, int C) { return (_Float16*)alloc((size_t)rows * C * 2); };
    _Float16* X0h = fbuf(M0 + 1, 32);
    _Float16* A0  = fbuf(M0, 32);
    _Float16* B0  = fbuf(M0 + 1, 32);
    _Float16* C0  = fbuf(M0, 32);
    _Float16* B1  = fbuf(M0 + 1, 32);
    _Float16* D0  = fbuf(M0, 32);      // becomes x0
    _Float16* B2  = fbuf(M0 + 1, 32);
    _Float16* Yb  = fbuf(M1, 32);
    _Float16* BY0 = fbuf(M1 + 1, 32);
    _Float16* CY  = fbuf(M1, 32);
    _Float16* BY1 = fbuf(M1 + 1, 32);
    _Float16* DY  = fbuf(M1, 32);      // becomes y0
    _Float16* B5  = fbuf(M1 + 1, 32);
    _Float16* Zb  = fbuf(M2, 32);
    _Float16* BZ0 = fbuf(M2 + 1, 32);
    _Float16* CZ  = fbuf(M2, 32);
    _Float16* BZ1 = fbuf(M2 + 1, 32);
    _Float16* DZ  = fbuf(M2, 32);      // becomes z0
    _Float16* Hraw = fbuf(M0, 96);
    _Float16* H    = fbuf(M0 + 1, 96);
    _Float16* XN   = fbuf(Mn, 32);

    auto cdiv = [](long long a, long long b) { return (unsigned)((a + b - 1) / b); };

    // zero the BN accumulators every call (graph-replay safe)
    hipMemsetAsync(stats, 0, 10 * 192 * sizeof(float), stream);

    // --- weight packing ---
    pack_w_kernel<<<cdiv(27LL * 1024, TPB), TPB, 0, stream>>>(W1, W1p, 27, 16, 1);
    for (int i = 0; i < 6; ++i)
        pack_w_kernel<<<cdiv(27LL * 1024, TPB), TPB, 0, stream>>>(
            Wres + (size_t)i * 27 * 32 * 32, Wresp[i], 27, 32, 1);
    for (int i = 0; i < 2; ++i)
        pack_w_kernel<<<cdiv(8LL * 1024, TPB), TPB, 0, stream>>>(
            Wdown + (size_t)i * 8 * 32 * 32, Wdownp[i], 8, 32, 1);
    pack_w_kernel<<<cdiv(27LL * 3 * 1024, TPB), TPB, 0, stream>>>(Wn1, Wn1p, 27, 96, 3);

    auto conv27 = [&](const _Float16* X, const _Float16* Bp, const int* nbr, _Float16* Out,
                      int Mout, int sentinel) {
        int blocks = (((Mout + 15) / 16) + 7) / 8;
        conv_wmma_kernel<1, 27><<<blocks, 256, 27 * 1024 * 2, stream>>>(
            X, Bp, nbr, Out, Mout, sentinel, 0);
    };
    auto conv8 = [&](const _Float16* X, const _Float16* Bp, const int* nbr, _Float16* Out,
                     int Mout, int sentinel) {
        int blocks = (((Mout + 15) / 16) + 7) / 8;
        conv_wmma_kernel<1, 8><<<blocks, 256, 8 * 1024 * 2, stream>>>(
            X, Bp, nbr, Out, Mout, sentinel, 0);
    };
    auto bnrelu = [&](const _Float16* X, _Float16* Y, float* slot,
                      const float* g, const float* b, int M, int C) {
        long long n = (long long)M * C;
        unsigned gs = cdiv(n, TPB); if (gs > 2048) gs = 2048;
        bn_stats_kernel<<<gs, TPB, 0, stream>>>(X, slot, M, C);
        bn_apply_kernel<<<cdiv((long long)(M + 1) * C, TPB), TPB, 0, stream>>>(X, Y, slot,
                                                                               g, b, M, C);
    };
    auto addf = [&](_Float16* Y, const _Float16* X, long long n) {
        add_kernel<<<cdiv(n, TPB), TPB, 0, stream>>>(Y, X, n);
    };

    // --- forward ---
    feats_to_h_kernel<<<cdiv((long long)(M0 + 1) * 32, TPB), TPB, 0, stream>>>(
        feats, X0h, M0, 16, 32);

    conv27(X0h, W1p, rb0, A0, M0, M0);                               // p1
    bnrelu(A0, B0, st(0), bn_g + 0, bn_b + 0, M0, 32);               // resblock L0
    conv27(B0, Wresp[0], rb0, C0, M0, M0);
    bnrelu(C0, B1, st(1), bn_g + 32, bn_b + 32, M0, 32);
    conv27(B1, Wresp[1], rb0, D0, M0, M0);
    addf(D0, A0, (long long)M0 * 32);                                // x0 = x + y

    bnrelu(D0, B2, st(2), bn_g + 64, bn_b + 64, M0, 32);             // down L0->L1
    conv8(B2, Wdownp[0], child0, Yb, M1, M0);

    bnrelu(Yb, BY0, st(3), bn_g + 96, bn_b + 96, M1, 32);            // resblock L1
    conv27(BY0, Wresp[2], rb1, CY, M1, M1);
    bnrelu(CY, BY1, st(4), bn_g + 128, bn_b + 128, M1, 32);
    conv27(BY1, Wresp[3], rb1, DY, M1, M1);
    addf(DY, Yb, (long long)M1 * 32);                                // y0

    bnrelu(DY, B5, st(5), bn_g + 160, bn_b + 160, M1, 32);           // down L1->L2
    conv8(B5, Wdownp[1], child1, Zb, M2, M1);

    bnrelu(Zb, BZ0, st(6), bn_g + 192, bn_b + 192, M2, 32);          // resblock L2
    conv27(BZ0, Wresp[4], rb2, CZ, M2, M2);
    bnrelu(CZ, BZ1, st(7), bn_g + 224, bn_b + 224, M2, 32);
    conv27(BZ1, Wresp[5], rb2, DZ, M2, M2);
    addf(DZ, Zb, (long long)M2 * 32);                                // z0

    concat3_kernel<<<cdiv((long long)M0 * 96, TPB), TPB, 0, stream>>>(
        D0, DY, DZ, parent0, parent1, Hraw, M0);                     // unpool + join
    bnrelu(Hraw, H, st(9), g3, b3, M0, 96);                          // p3 (3nf)

    // n1: big conv over 8x-repeated rows; shift=3 reads h[row>>3] (no repeat)
    {
        int blocks = (((Mn + 15) / 16) + 7) / 8;
        conv_wmma_kernel<3, 27><<<blocks, 256, 3 * 27 * 1024 * 2, stream>>>(
            H, Wn1p, rbn, XN, Mn, Mn, 3);
    }

    {   // n2 BN stats + heads
        long long n = (long long)Mn * 32;
        unsigned gs = cdiv(n, TPB); if (gs > 2048) gs = 2048;
        bn_stats_kernel<<<gs, TPB, 0, stream>>>(XN, st(8), Mn, 32);
    }
    head_kernel<<<cdiv((long long)Mn, TPB), TPB, 0, stream>>>(
        XN, st(8), bn_g + 256, bn_b + 256, Wlin, blin, Wsdf, bsdf, locs,
        (float*)d_out, Mn);
}
